// StreamingTransformer_40922448397087
// MI455X (gfx1250) — matrix operand order repaired and checked by reading, not compiled, not run on hardware
//
#include <hip/hip_runtime.h>
#include <hip/hip_bf16.h>
#include <cstdint>

// ---------------------------------------------------------------------------
// Types / helpers for CDNA5 WMMA (wave32, 16x16x32 bf16 -> f32)
// ---------------------------------------------------------------------------
typedef __attribute__((ext_vector_type(16))) __bf16 v16bf;
typedef __attribute__((ext_vector_type(8)))  float  v8f;

union FragU { uint4 u[2]; v16bf v; };

static __device__ __forceinline__ v16bf mkfrag(uint4 lo, uint4 hi) {
    FragU t; t.u[0] = lo; t.u[1] = hi; return t.v;
}

static __device__ __forceinline__ v8f wmma_bf16(v16bf a, v16bf b, v8f c) {
    // (neg_a, A, neg_b, B, c_mod, C, reuse_a, reuse_b)
    return __builtin_amdgcn_wmma_f32_16x16x32_bf16(false, a, false, b, (short)0, c, false, false);
}

// gfx1250 async copy: 16B per lane, global -> LDS, tracked on ASYNCcnt.
static __device__ __forceinline__ void async_b128(const void* gptr, void* lptr) {
    unsigned lds = (unsigned)(uintptr_t)lptr;              // low 32b of flat = LDS offset
    unsigned long long ga = (unsigned long long)(uintptr_t)gptr;
    asm volatile("global_load_async_to_lds_b128 %0, %1, off"
                 :: "v"(lds), "v"(ga) : "memory");
}

#define BN_ 8
#define SN_ 1000
#define SPAD_ 1024
#define CN_ 512
#define HN_ 8
#define DN_ 64

// ---------------------------------------------------------------------------
// 1) LayerNorm (layer 5) + fp32 -> bf16
// ---------------------------------------------------------------------------
__global__ __launch_bounds__(256) void ln_bf16_kernel(
    const float* __restrict__ x, const float* __restrict__ g,
    const float* __restrict__ be, __bf16* __restrict__ xn)
{
    const int row = blockIdx.x;
    const int tid = threadIdx.x;
    const float* xr = x + (size_t)row * CN_;
    float v0 = xr[tid];
    float v1 = xr[tid + 256];
    __shared__ float red[8];

    float s = v0 + v1;
#pragma unroll
    for (int o = 16; o > 0; o >>= 1) s += __shfl_xor(s, o, 32);
    if ((tid & 31) == 0) red[tid >> 5] = s;
    __syncthreads();
    float tot = 0.f;
#pragma unroll
    for (int i = 0; i < 8; i++) tot += red[i];
    const float mean = tot * (1.0f / 512.0f);
    __syncthreads();

    float d0 = v0 - mean, d1 = v1 - mean;
    float vs = d0 * d0 + d1 * d1;
#pragma unroll
    for (int o = 16; o > 0; o >>= 1) vs += __shfl_xor(vs, o, 32);
    if ((tid & 31) == 0) red[tid >> 5] = vs;
    __syncthreads();
    float vtot = 0.f;
#pragma unroll
    for (int i = 0; i < 8; i++) vtot += red[i];
    const float rstd = rsqrtf(vtot * (1.0f / 512.0f) + 1e-5f);

    xn[(size_t)row * CN_ + tid]       = (__bf16)(d0 * rstd * g[tid] + be[tid]);
    xn[(size_t)row * CN_ + tid + 256] = (__bf16)(d1 * rstd * g[tid + 256] + be[tid + 256]);
}

// ---------------------------------------------------------------------------
// 2) zero fill (padded q/k/vT buffers)
// ---------------------------------------------------------------------------
__global__ __launch_bounds__(256) void zero_kernel(unsigned int* __restrict__ p, unsigned int n) {
    unsigned int i = blockIdx.x * blockDim.x + threadIdx.x;
    const unsigned int stride = gridDim.x * blockDim.x;
    for (; i < n; i += stride) p[i] = 0u;
}

// ---------------------------------------------------------------------------
// 3) one-shot f32 -> bf16 weight conversion (so GEMM tiles are async-copyable)
// ---------------------------------------------------------------------------
__global__ __launch_bounds__(256) void cvt_bf16_kernel(
    const float* __restrict__ src, __bf16* __restrict__ dst, unsigned n)
{
    unsigned i = (blockIdx.x * blockDim.x + threadIdx.x) * 4u;
    const unsigned stride = gridDim.x * blockDim.x * 4u;
    for (; i < n; i += stride) {
        float4 f = *(const float4*)(src + i);
        union { __bf16 t[4]; uint2 u; } cv;
        cv.t[0] = (__bf16)f.x; cv.t[1] = (__bf16)f.y;
        cv.t[2] = (__bf16)f.z; cv.t[3] = (__bf16)f.w;
        *(uint2*)(dst + i) = cv.u;
    }
}

// ---------------------------------------------------------------------------
// 4) GEMM: C[M,512] = A_bf16[M,512] @ Wb_bf16[512,512]^T + bias
//    Double-buffered LDS tiles filled by global_load_async_to_lds_b128.
//    mode 0: scatter bf16 -> q/k layout [B,H,Spad,D]
//    mode 1: scatter bf16 -> vT layout  [B,H,D,Spad]
//    mode 2: f32 row-major [M,512] (final output)
// Block tile 128x128, K-step 32, 8 waves (4x2), wave tile 32x64.
// ---------------------------------------------------------------------------
__global__ __launch_bounds__(256) void gemm_bf16_kernel(
    const __bf16* __restrict__ A, const __bf16* __restrict__ Wb,
    const float* __restrict__ bias, void* __restrict__ out,
    int M, int S, int Spad, int mode)
{
    const int K = 512, N = 512;
    __shared__ __attribute__((aligned(16))) __bf16 lA[2][128 * 32];
    __shared__ __attribute__((aligned(16))) __bf16 lW[2][128 * 32];

    const int bm = blockIdx.x * 128;
    const int bn = blockIdx.y * 128;
    const int tid = threadIdx.x;
    const int lane = tid & 31, wave = tid >> 5;
    const int wm = wave & 3, wn = wave >> 2;     // 4 x 2 wave grid
    const int hi16 = lane >> 4, l16 = lane & 15;

    v8f acc[2][4];
    const v8f vzero = {};
#pragma unroll
    for (int mi = 0; mi < 2; mi++)
#pragma unroll
        for (int ni = 0; ni < 4; ni++) acc[mi][ni] = vzero;

    // per-thread 32B chunk of each 128x32 tile
    const int tr = tid >> 1, tc = (tid & 1) * 16;
    int gmA = bm + tr; if (gmA >= M) gmA = M - 1;   // clamp: dead rows discarded in epilogue
    const int gnW = bn + tr;

    auto issueTile = [&](int k0, int bi) {
        const __bf16* ga = A  + (size_t)gmA * K + k0 + tc;
        __bf16*       la = &lA[bi][tr * 32 + tc];
        async_b128(ga,     la);
        async_b128(ga + 8, la + 8);
        const __bf16* gw = Wb + (size_t)gnW * K + k0 + tc;
        __bf16*       lw = &lW[bi][tr * 32 + tc];
        async_b128(gw,     lw);
        async_b128(gw + 8, lw + 8);
    };

    const int NK = K / 32;           // 16
    issueTile(0, 0);
    for (int it = 0; it < NK; ++it) {
        const int bi = it & 1;
        if (it + 1 < NK) {
            issueTile((it + 1) * 32, (it + 1) & 1);
            asm volatile("s_wait_asynccnt 0x4" ::: "memory");   // current tile landed
        } else {
            asm volatile("s_wait_asynccnt 0x0" ::: "memory");
        }
        __syncthreads();

        v16bf af[2], bfr[4];
#pragma unroll
        for (int mi = 0; mi < 2; mi++) {
            const int r = wm * 32 + mi * 16 + l16;
            const int kc = hi16 ? 8 : 0;                 // A layout: K halves split per lane half
            const uint4* p = (const uint4*)&lA[bi][r * 32 + kc];
            af[mi] = mkfrag(p[0], p[2]);                 // K = kc..kc+7 , kc+16..kc+23
        }
#pragma unroll
        for (int ni = 0; ni < 4; ni++) {
            const int r = wn * 64 + ni * 16 + l16;       // row of W == output col N
            const int kc = hi16 ? 8 : 0;                 // B layout: K halves split per lane half
            const uint4* p = (const uint4*)&lW[bi][r * 32 + kc];
            bfr[ni] = mkfrag(p[0], p[2]);
        }
#pragma unroll
        for (int mi = 0; mi < 2; mi++)
#pragma unroll
            for (int ni = 0; ni < 4; ni++)
                acc[mi][ni] = wmma_bf16(af[mi], bfr[ni], acc[mi][ni]);
        __syncthreads();              // all waves done reading before buffer reuse
    }

    // --- epilogue ---
#pragma unroll
    for (int mi = 0; mi < 2; mi++)
#pragma unroll
        for (int ni = 0; ni < 4; ni++) {
            const int n = bn + wn * 64 + ni * 16 + l16;
            const float bv = bias[n];
#pragma unroll
            for (int v = 0; v < 8; v++) {
                const int m = bm + wm * 32 + mi * 16 + v + 8 * hi16;
                if (m >= M) continue;
                const float val = acc[mi][ni][v] + bv;
                if (mode == 2) {
                    ((float*)out)[(size_t)m * N + n] = val;
                } else {
                    const int b = m / S, s = m - b * S;
                    const int h = n >> 6, d = n & 63;
                    const __bf16 bw = (__bf16)val;
                    if (mode == 0)
                        ((__bf16*)out)[(((size_t)b * HN_ + h) * Spad + s) * DN_ + d] = bw;
                    else
                        ((__bf16*)out)[(((size_t)b * HN_ + h) * DN_ + d) * Spad + s] = bw;
                }
            }
        }
}

// ---------------------------------------------------------------------------
// 5) Flash attention per (b,h): scores = qk^T * 0.125 + (i-j); softmax; @ V
//    grid: B*H*(Spad/128) blocks, 8 waves, each wave = 16 query rows.
//    Shared K/V tiles staged once per block via async-to-LDS (double buffered).
// ---------------------------------------------------------------------------
__global__ __launch_bounds__(256) void attn_kernel(
    const __bf16* __restrict__ q, const __bf16* __restrict__ k,
    const __bf16* __restrict__ vt, __bf16* __restrict__ ctx,
    int S, int Spad)
{
    const int qb = blockIdx.x & 7;
    const int bh = blockIdx.x >> 3;
    const int b = bh >> 3, h = bh & 7;
    const int tid = threadIdx.x;
    const int lane = tid & 31, wave = tid >> 5;
    const int hi16 = lane >> 4, l16 = lane & 15;
    const int qBase = qb * 128 + wave * 16;

    __shared__ __attribute__((aligned(16))) __bf16 lK[2][32 * 64];  // keys x d
    __shared__ __attribute__((aligned(16))) __bf16 lV[2][64 * 32];  // d x keys
    __shared__ __attribute__((aligned(16))) __bf16 lP[8][16 * 32];  // per-wave P staging

    const __bf16* qp = q  + (size_t)bh * Spad * DN_;
    const __bf16* kp = k  + (size_t)bh * Spad * DN_;
    const __bf16* vp = vt + (size_t)bh * DN_ * Spad;

    // q A-fragments (16 rows x 64 K), loaded once
    v16bf aq[2];
    {
        const __bf16* row = qp + (size_t)(qBase + l16) * DN_;
        const int d0 = hi16 ? 8 : 0;
        const uint4* p0 = (const uint4*)(row + d0);
        aq[0] = mkfrag(p0[0], p0[2]);
        const uint4* p1 = (const uint4*)(row + 32 + d0);
        aq[1] = mkfrag(p1[0], p1[2]);
    }

    float mrow[8], lrow[8];
    v8f o[4];
    const v8f vzero = {};
#pragma unroll
    for (int v = 0; v < 8; v++) { mrow[v] = -1e30f; lrow[v] = 0.f; }
#pragma unroll
    for (int dt = 0; dt < 4; dt++) o[dt] = vzero;

    const float scale = 0.125f;     // 1/sqrt(64)
    __bf16* Pw = lP[wave];

    // cooperative K/V tile staging (one 16B async chunk per thread per tile)
    const int krow_ = tid >> 3, kcol_ = (tid & 7) * 8;   // 32 rows x 64 cols
    const int vrow_ = tid >> 2, vcol_ = (tid & 3) * 8;   // 64 rows x 32 cols
    auto issueKV = [&](int kt, int bi) {
        async_b128(kp + (size_t)(kt + krow_) * DN_ + kcol_, &lK[bi][krow_ * 64 + kcol_]);
        async_b128(vp + (size_t)vrow_ * Spad + kt + vcol_,  &lV[bi][vrow_ * 32 + vcol_]);
    };

    const int NT = Spad / 32;        // 32
    issueKV(0, 0);
    for (int it = 0; it < NT; ++it) {
        const int bi = it & 1;
        const int kt = it * 32;
        if (it + 1 < NT) {
            issueKV((it + 1) * 32, (it + 1) & 1);
            asm volatile("s_wait_asynccnt 0x2" ::: "memory");
        } else {
            asm volatile("s_wait_asynccnt 0x0" ::: "memory");
        }
        __syncthreads();

        // ---- scores: two 16-key N-tiles, chained over D (2x k=32) ----
        v8f sc[2];
#pragma unroll
        for (int nt = 0; nt < 2; nt++) {
            const int d0 = hi16 ? 8 : 0;
            const uint4* p0 = (const uint4*)&lK[bi][(nt * 16 + l16) * 64 + d0];
            v16bf b0 = mkfrag(p0[0], p0[2]);
            const uint4* p1 = (const uint4*)&lK[bi][(nt * 16 + l16) * 64 + 32 + d0];
            v16bf b1 = mkfrag(p1[0], p1[2]);
            sc[nt] = wmma_bf16(aq[0], b0, vzero);
            sc[nt] = wmma_bf16(aq[1], b1, sc[nt]);
        }
        // ---- scale + relative-position bias + key mask, row max ----
        float tmax[8];
#pragma unroll
        for (int v = 0; v < 8; v++) {
            const int qi = qBase + v + 8 * hi16;
            const int kj0 = kt + l16;
            const int kj1 = kt + 16 + l16;
            float s0 = sc[0][v] * scale + (float)(qi - kj0);
            float s1 = sc[1][v] * scale + (float)(qi - kj1);
            if (kj0 >= S) s0 = -3.0e38f;
            if (kj1 >= S) s1 = -3.0e38f;
            sc[0][v] = s0; sc[1][v] = s1;
            float mx = fmaxf(s0, s1);
#pragma unroll
            for (int off = 8; off > 0; off >>= 1) mx = fmaxf(mx, __shfl_xor(mx, off, 32));
            tmax[v] = mx;
        }
        // ---- online softmax update ----
        float alpha[8];
#pragma unroll
        for (int v = 0; v < 8; v++) {
            const float nm = fmaxf(mrow[v], tmax[v]);
            alpha[v] = __expf(mrow[v] - nm);
            mrow[v] = nm;
        }
#pragma unroll
        for (int v = 0; v < 8; v++) {
            const float p0 = __expf(sc[0][v] - mrow[v]);
            const float p1 = __expf(sc[1][v] - mrow[v]);
            const int r = v + 8 * hi16;
            Pw[r * 32 + l16]      = (__bf16)p0;      // C-layout -> LDS
            Pw[r * 32 + 16 + l16] = (__bf16)p1;
            float rs = p0 + p1;
#pragma unroll
            for (int off = 8; off > 0; off >>= 1) rs += __shfl_xor(rs, off, 32);
            lrow[v] = lrow[v] * alpha[v] + rs;
        }
#pragma unroll
        for (int dt = 0; dt < 4; dt++)
#pragma unroll
            for (int v = 0; v < 8; v++) o[dt][v] *= alpha[v];

        // same-wave LDS RAW: hardware in-order via DScnt; fence the compiler too
        asm volatile("s_wait_dscnt 0" ::: "memory");

        // ---- P back as A-fragment (16 x 32) ----
        v16bf pa;
        {
            const int kc = hi16 ? 8 : 0;
            const uint4* p = (const uint4*)&Pw[l16 * 32 + kc];
            pa = mkfrag(p[0], p[2]);
        }
        // ---- ctx += P @ V (LDS vT tile gives contiguous B-operand) ----
#pragma unroll
        for (int dt = 0; dt < 4; dt++) {
            const uint4* p = (const uint4*)&lV[bi][(dt * 16 + l16) * 32 + (hi16 ? 8 : 0)];
            v16bf bv = mkfrag(p[0], p[2]);
            o[dt] = wmma_bf16(pa, bv, o[dt]);
        }
        __syncthreads();             // all waves done with lK/lV[bi] before refill
    }

    // ---- normalize + write ctx [B,S, H*64] bf16 ----
#pragma unroll
    for (int dt = 0; dt < 4; dt++)
#pragma unroll
        for (int v = 0; v < 8; v++) {
            const int qi = qBase + v + 8 * hi16;
            if (qi < S) {
                const float val = o[dt][v] / lrow[v];
                ctx[((size_t)b * S + qi) * CN_ + h * DN_ + dt * 16 + l16] = (__bf16)val;
            }
        }
}

// ---------------------------------------------------------------------------
// Launch: only layer L-1 = 5 matters (x is never updated between layers).
// ---------------------------------------------------------------------------
extern "C" void kernel_launch(void* const* d_in, const int* in_sizes, int n_in,
                              void* d_out, int out_size, void* d_ws, size_t ws_size,
                              hipStream_t stream) {
    (void)in_sizes; (void)n_in; (void)out_size; (void)ws_size;
    const float* x    = (const float*)d_in[0];
    const float* ln_g = (const float*)d_in[1];
    const float* ln_b = (const float*)d_in[2];
    const float* Wq   = (const float*)d_in[3];
    const float* bq   = (const float*)d_in[4];
    const float* Wk   = (const float*)d_in[5];
    const float* bk   = (const float*)d_in[6];
    const float* Wv   = (const float*)d_in[7];
    const float* bv   = (const float*)d_in[8];
    const float* Wo   = (const float*)d_in[9];
    const float* bo   = (const float*)d_in[10];
    float* out = (float*)d_out;

    const int LAYER = 5;
    const size_t WOFF = (size_t)LAYER * 512 * 512;
    const size_t VOFF = (size_t)LAYER * 512;

    const size_t XN_BYTES  = (size_t)8000 * 512 * 2;            // 8,192,000
    const size_t QKV_BYTES = (size_t)8 * 8 * SPAD_ * DN_ * 2;   // 8,388,608 each
    const size_t CTX_BYTES = (size_t)8000 * 512 * 2 + 65536;    // + tail slack for clamp reads
    const size_t WB_BYTES  = (size_t)512 * 512 * 2;             // 524,288 each

    char* ws = (char*)d_ws;
    __bf16* xn  = (__bf16*)ws;
    __bf16* qb_ = (__bf16*)(ws + XN_BYTES);
    __bf16* kb_ = (__bf16*)(ws + XN_BYTES + QKV_BYTES);
    __bf16* vtb = (__bf16*)(ws + XN_BYTES + 2 * QKV_BYTES);
    __bf16* ctx = (__bf16*)(ws + XN_BYTES + 3 * QKV_BYTES);
    __bf16* wqb = (__bf16*)(ws + XN_BYTES + 3 * QKV_BYTES + CTX_BYTES);
    __bf16* wkb = wqb + 512 * 512;
    __bf16* wvb = wkb + 512 * 512;
    __bf16* wob = wvb + 512 * 512;
    (void)WB_BYTES;

    // 1) LN(x) -> bf16
    ln_bf16_kernel<<<8000, 256, 0, stream>>>(x, ln_g + VOFF, ln_b + VOFF, xn);
    // 2) zero padded q/k/vT buffers (contiguous region)
    zero_kernel<<<2048, 256, 0, stream>>>((unsigned int*)qb_,
                                          (unsigned int)(3 * QKV_BYTES / 4));
    // 3) weights f32 -> bf16 (one shot)
    cvt_bf16_kernel<<<256, 256, 0, stream>>>(Wq + WOFF, wqb, 512u * 512u);
    cvt_bf16_kernel<<<256, 256, 0, stream>>>(Wk + WOFF, wkb, 512u * 512u);
    cvt_bf16_kernel<<<256, 256, 0, stream>>>(Wv + WOFF, wvb, 512u * 512u);
    cvt_bf16_kernel<<<256, 256, 0, stream>>>(Wo + WOFF, wob, 512u * 512u);
    // 4-6) projections
    dim3 gg((8000 + 127) / 128, 4);
    gemm_bf16_kernel<<<gg, 256, 0, stream>>>(xn, wqb, bq + VOFF, (void*)qb_, 8000, SN_, SPAD_, 0);
    gemm_bf16_kernel<<<gg, 256, 0, stream>>>(xn, wkb, bk + VOFF, (void*)kb_, 8000, SN_, SPAD_, 0);
    gemm_bf16_kernel<<<gg, 256, 0, stream>>>(xn, wvb, bv + VOFF, (void*)vtb, 8000, SN_, SPAD_, 1);
    // 7) attention
    attn_kernel<<<8 * 8 * (SPAD_ / 128), 256, 0, stream>>>(qb_, kb_, vtb, ctx, SN_, SPAD_);
    // 8) output projection -> f32 d_out
    gemm_bf16_kernel<<<gg, 256, 0, stream>>>(ctx, wob, bo + VOFF, (void*)out, 8000, SN_, SPAD_, 2);
}
